// PointNetSAModule_85358180040891
// MI455X (gfx1250) — matrix operand order repaired and checked by reading, not compiled, not run on hardware
//
#include <hip/hip_runtime.h>
#include <hip/hip_bf16.h>
#include <cmath>

typedef __attribute__((ext_vector_type(16))) _Float16 v16h;
typedef __attribute__((ext_vector_type(2)))  _Float16 v2h;
typedef __attribute__((ext_vector_type(8)))  float    v8f;

#define NPTS  8192
#define BATCH 8
#define MCTR  1024
#define KNB   64
#define R2    0.04f
#define LDX   104   // padded stride (halves) for sXT: 52*n mod 64 bijective over 16 lanes
#define LDH   72    // padded stride (halves) for sH1T: 36*n mod 64 bijective over 16 lanes

// ---------------- WMMA helpers (CDNA5 16x16x32 f16 -> f32) ----------------
static __device__ __forceinline__ v8f wmma_f16(v16h a, v16h b, v8f c) {
  return __builtin_amdgcn_wmma_f32_16x16x32_f16(false, a, false, b, (short)0, c,
                                                false, false);
}

// A fragment: pre-swizzled weights, one contiguous 32B load per lane (32B aligned)
static __device__ __forceinline__ v16h load_a_frag(const _Float16* __restrict__ Wf,
                                                   int frag, int lane) {
  const _Float16* p = Wf + (frag * 32 + lane) * 16;
  v16h a;
  __builtin_memcpy(&a, __builtin_assume_aligned(p, 32), 32);
  return a;
}

// B fragment from transposed LDS tile S[n*ldc + c]: contiguous 32B per lane (16B aligned)
static __device__ __forceinline__ v16h load_b_fragT(const _Float16* S, int n0, int c0,
                                                    int ldc, int lane) {
  int n = lane & 15, hi = lane >> 4;
  const _Float16* p = S + (n0 + n) * ldc + c0 + 16 * hi;
  v16h b;
  __builtin_memcpy(&b, __builtin_assume_aligned(p, 16), 32);
  return b;
}

static __device__ __forceinline__ float silu_f(float x) {
  return x / (1.f + __expf(-x));
}

// ---------------- weight pre-swizzle into per-lane fragment order ----------------
// fragment element h of lane: K = 16*hi+h, row = lane&15
__global__ void prep_kernel(const float* __restrict__ w1, const float* __restrict__ w2,
                            _Float16* __restrict__ w1f, _Float16* __restrict__ w2f) {
  int t = blockIdx.x * 256 + threadIdx.x;
  int stride = gridDim.x * 256;
  for (int e = t; e < 4 * 3 * 32 * 16; e += stride) {   // w1: 4 strips x 3 chunks
    int h = e & 15, lanef = (e >> 4) & 31, f = e >> 9;
    int s = f / 3, cc = f % 3;
    int m = lanef & 15, hi = lanef >> 4;
    int o = 16 * s + m;
    int c = 32 * cc + 16 * hi + h;
    w1f[e] = (c < 67) ? (_Float16)w1[o * 67 + c] : (_Float16)0.f;
  }
  for (int e = t; e < 8 * 2 * 32 * 16; e += stride) {   // w2: 8 strips x 2 chunks
    int h = e & 15, lanef = (e >> 4) & 31, f = e >> 9;
    int s = f >> 1, cc = f & 1;
    int m = lanef & 15, hi = lanef >> 4;
    int o = 16 * s + m;
    int c = 32 * cc + 16 * hi + h;
    w2f[e] = (_Float16)w2[o * 64 + c];
  }
}

// ---------------- FPS: one block per batch ----------------
__global__ __launch_bounds__(1024) void fps_kernel(const float* __restrict__ coords,
                                                   int* __restrict__ c_idx) {
  __shared__ float wv_v[32];
  __shared__ int   wv_n[32];
  __shared__ int   s_last;
  int b = blockIdx.x, tid = threadIdx.x, lane = tid & 31, w = tid >> 5;
  const float* cb = coords + (size_t)b * 3 * NPTS;
  float px[8], py[8], pz[8], dist[8];
#pragma unroll
  for (int i = 0; i < 8; ++i) {
    int n = tid + i * 1024;
    px[i] = cb[n]; py[i] = cb[NPTS + n]; pz[i] = cb[2 * NPTS + n];
    dist[i] = 1e10f;
  }
  if (tid == 0) { s_last = 0; c_idx[b * MCTR] = 0; }
  __syncthreads();
  for (int step = 1; step < MCTR; ++step) {
    int last = s_last;
    float lx = cb[last], ly = cb[NPTS + last], lz = cb[2 * NPTS + last];
    float bv = -1.f; int bn = 0;
#pragma unroll
    for (int i = 0; i < 8; ++i) {
      float dx = px[i] - lx, dy = py[i] - ly, dz = pz[i] - lz;
      float d = fminf(dist[i], dx * dx + dy * dy + dz * dz);
      dist[i] = d;
      if (d > bv) { bv = d; bn = tid + i * 1024; }
    }
#pragma unroll
    for (int off = 16; off >= 1; off >>= 1) {            // argmax, lowest-index ties
      float ov = __shfl_xor(bv, off, 32);
      int   on = __shfl_xor(bn, off, 32);
      if (ov > bv || (ov == bv && on < bn)) { bv = ov; bn = on; }
    }
    if (lane == 0) { wv_v[w] = bv; wv_n[w] = bn; }
    __syncthreads();
    if (w == 0) {
      bv = wv_v[lane]; bn = wv_n[lane];
#pragma unroll
      for (int off = 16; off >= 1; off >>= 1) {
        float ov = __shfl_xor(bv, off, 32);
        int   on = __shfl_xor(bn, off, 32);
        if (ov > bv || (ov == bv && on < bn)) { bv = ov; bn = on; }
      }
      if (lane == 0) { s_last = bn; c_idx[b * MCTR + step] = bn; }
    }
    __syncthreads();
  }
}

// ---------------- centers -> d_out section 2 ----------------
__global__ void centers_kernel(const float* __restrict__ coords, const int* __restrict__ c_idx,
                               float* __restrict__ outc) {
  int t = blockIdx.x * 256 + threadIdx.x;
  if (t >= BATCH * MCTR) return;
  int b = t >> 10, m = t & 1023;
  int cid = c_idx[t];
#pragma unroll
  for (int d = 0; d < 3; ++d)
    outc[(size_t)b * 3 * MCTR + d * MCTR + m] = coords[(size_t)b * 3 * NPTS + d * NPTS + cid];
}

// ---------------- ball query: one wave per center ----------------
__global__ __launch_bounds__(256) void ballq_kernel(const float* __restrict__ coords,
                                                    const int* __restrict__ c_idx,
                                                    int* __restrict__ idx) {
  int gw = blockIdx.x * 8 + (threadIdx.x >> 5);
  int lane = threadIdx.x & 31;
  int b = gw >> 10;
  const float* cb = coords + (size_t)b * 3 * NPTS;
  int cid = c_idx[gw];
  float cx = cb[cid], cy = cb[NPTS + cid], cz = cb[2 * NPTS + cid];
  int* ob = idx + (size_t)gw * KNB;
  int count = 0, first_n = 0;
  for (int base = 0; base < NPTS && count < KNB; base += 32) {
    if (base + 64 < NPTS) __builtin_prefetch(cb + base + 32 + lane, 0, 1);  // global_prefetch_b8
    int n = base + lane;
    float dx = cb[n] - cx, dy = cb[NPTS + n] - cy, dz = cb[2 * NPTS + n] - cz;
    bool pred = (dx * dx + dy * dy + dz * dz) < R2;
    unsigned mask = (unsigned)__ballot(pred);
    if (count == 0 && mask) first_n = base + (__ffs(mask) - 1);
    int pos = count + __popc(mask & ((1u << lane) - 1));
    if (pred && pos < KNB) ob[pos] = n;
    count += __popc(mask);
  }
  if (count < KNB) {
    int fill = count ? first_n : 0;
    for (int p = count + lane; p < KNB; p += 32) ob[p] = fill;
  }
}

// ---------------- temb gather + max over K -> d_out section 3 ----------------
__global__ __launch_bounds__(64) void temb_kernel(const float* __restrict__ temb,
                                                  const int* __restrict__ idx,
                                                  float* __restrict__ out_t) {
  __shared__ int s_idx[KNB];
  int bm = blockIdx.x, b = bm >> 10, m = bm & 1023, c = threadIdx.x;
  s_idx[c] = idx[(size_t)bm * KNB + c];
  __syncthreads();
  const float* tb = temb + (size_t)b * 64 * NPTS + (size_t)c * NPTS;
  float mx = tb[s_idx[0]];
#pragma unroll 8
  for (int k = 1; k < KNB; ++k) mx = fmaxf(mx, tb[s_idx[k]]);
  out_t[(size_t)b * 64 * MCTR + c * MCTR + m] = mx;
}

// ---------------- finalize group stats ----------------
__global__ void finalize_kernel(const float* __restrict__ ps, const float* __restrict__ pss,
                                float* __restrict__ nrm, float inv_cnt) {
  int t = threadIdx.x;  // 64 = B * NUM_GROUPS
  float s = 0.f, ss = 0.f;
  for (int m = 0; m < MCTR; ++m) { s += ps[(t << 10) + m]; ss += pss[(t << 10) + m]; }
  float mu = s * inv_cnt;
  float var = ss * inv_cnt - mu * mu;
  nrm[t * 2] = mu;
  nrm[t * 2 + 1] = rsqrtf(var + 1e-5f);
}

// ---------------- fused per-center WMMA MLP (3 passes) ----------------
template <int PASS>
__global__ __launch_bounds__(128) void mlp_kernel(
    const float* __restrict__ features, const float* __restrict__ coords,
    const int* __restrict__ idx, const int* __restrict__ c_idx,
    const _Float16* __restrict__ w1f, const _Float16* __restrict__ w2f,
    const float* __restrict__ b1, const float* __restrict__ gw1, const float* __restrict__ gb1,
    const float* __restrict__ b2, const float* __restrict__ gw2, const float* __restrict__ gb2,
    const float* __restrict__ norm1, const float* __restrict__ norm2,
    float* __restrict__ ps1, float* __restrict__ pss1,
    float* __restrict__ ps2, float* __restrict__ pss2,
    float* __restrict__ out) {
  __shared__ _Float16 sXT[64 * LDX];   // transposed gathered input: [n][c], c padded to 96
  __shared__ _Float16 sH1T[64 * LDH];  // transposed layer-1 activations: [n][o]
  __shared__ int   s_idx[KNB];
  __shared__ float s_cc[3];

  int bm = blockIdx.x, b = bm >> 10, m = bm & 1023;
  int tid = threadIdx.x, lane = tid & 31, wv = tid >> 5;
  int hi = lane >> 4, l16 = lane & 15;

  if (tid < KNB) s_idx[tid] = idx[(size_t)bm * KNB + tid];
  if (tid < 3) {
    int cid = c_idx[bm];
    s_cc[tid] = coords[(size_t)b * 3 * NPTS + tid * NPTS + cid];
  }
  __syncthreads();
  // gather X transposed, two consecutive channels per 32-bit LDS store
  for (int e = tid; e < 48 * 64; e += 128) {
    int r2 = e >> 6, k = e & 63;
    int c0 = r2 * 2;
    float v0 = 0.f, v1 = 0.f;
    int id = s_idx[k];
#pragma unroll
    for (int u = 0; u < 2; ++u) {
      int c = c0 + u;
      float v = 0.f;
      if (c < 3)       v = coords[(size_t)b * 3 * NPTS + c * NPTS + id] - s_cc[c];
      else if (c < 67) v = features[(size_t)b * 64 * NPTS + (c - 3) * NPTS + id];
      if (u == 0) v0 = v; else v1 = v;
    }
    *(v2h*)(sXT + k * LDX + c0) = (v2h){(_Float16)v0, (_Float16)v1};
  }
  __syncthreads();

  // ---- Layer 1: wave wv owns output rows [16wv, 16wv+16) ----
  int o0 = wv * 16;
  v8f binit;
#pragma unroll
  for (int j = 0; j < 8; ++j) binit[j] = b1[o0 + j + 8 * hi];
  v8f acc[4];
#pragma unroll
  for (int t = 0; t < 4; ++t) acc[t] = binit;
#pragma unroll
  for (int cc = 0; cc < 3; ++cc) {
    v16h a = load_a_frag(w1f, wv * 3 + cc, lane);
#pragma unroll
    for (int t = 0; t < 4; ++t) {
      v16h bf = load_b_fragT(sXT, t * 16, 32 * cc, LDX, lane);
      acc[t] = wmma_f16(a, bf, acc[t]);
    }
  }

  if (PASS == 1) {  // per-(b,group) partial sums; group = 2*wv + hi (8 rows/group)
    float s = 0.f, ss = 0.f;
#pragma unroll
    for (int t = 0; t < 4; ++t)
#pragma unroll
      for (int j = 0; j < 8; ++j) { float v = acc[t][j]; s += v; ss += v * v; }
#pragma unroll
    for (int off = 8; off >= 1; off >>= 1) { s += __shfl_xor(s, off, 32); ss += __shfl_xor(ss, off, 32); }
    if (l16 == 0) {
      int g = 2 * wv + hi;
      ps1 [((b * 8 + g) << 10) + m] = s;
      pss1[((b * 8 + g) << 10) + m] = ss;
    }
    return;
  }

  // ---- normalize + SiLU -> sH1T (two consecutive o per 32-bit store) ----
  {
    int g = 2 * wv + hi;
    float mu = norm1[(b * 8 + g) * 2], rs = norm1[(b * 8 + g) * 2 + 1];
#pragma unroll
    for (int t = 0; t < 4; ++t) {
      int n = t * 16 + l16;
#pragma unroll
      for (int j = 0; j < 8; j += 2) {
        int o = o0 + j + 8 * hi;
        float h0 = (acc[t][j]     - mu) * rs * gw1[o]     + gb1[o];
        float h1 = (acc[t][j + 1] - mu) * rs * gw1[o + 1] + gb1[o + 1];
        *(v2h*)(sH1T + n * LDH + o) = (v2h){(_Float16)silu_f(h0), (_Float16)silu_f(h1)};
      }
    }
  }
  __syncthreads();

  // ---- Layer 2: wave wv owns strips 16wv and 64+16wv (each strip = one group of 16) ----
#pragma unroll
  for (int half = 0; half < 2; ++half) {
    int s2i = half * 4 + wv;          // strip index 0..7
    int O0 = s2i * 16;
    v8f bi2;
#pragma unroll
    for (int j = 0; j < 8; ++j) bi2[j] = b2[O0 + j + 8 * hi];
    v8f acc2[4];
#pragma unroll
    for (int t = 0; t < 4; ++t) acc2[t] = bi2;
#pragma unroll
    for (int cc = 0; cc < 2; ++cc) {
      v16h a = load_a_frag(w2f, s2i * 2 + cc, lane);
#pragma unroll
      for (int t = 0; t < 4; ++t) {
        v16h bf = load_b_fragT(sH1T, t * 16, 32 * cc, LDH, lane);
        acc2[t] = wmma_f16(a, bf, acc2[t]);
      }
    }
    int g = s2i;
    if (PASS == 2) {
      float s = 0.f, ss = 0.f;
#pragma unroll
      for (int t = 0; t < 4; ++t)
#pragma unroll
        for (int j = 0; j < 8; ++j) { float v = acc2[t][j]; s += v; ss += v * v; }
#pragma unroll
      for (int off = 16; off >= 1; off >>= 1) { s += __shfl_xor(s, off, 32); ss += __shfl_xor(ss, off, 32); }
      if (lane == 0) {
        ps2 [((b * 8 + g) << 10) + m] = s;
        pss2[((b * 8 + g) << 10) + m] = ss;
      }
    } else {  // PASS == 3: normalize + SiLU + max over K -> out
      float mu = norm2[(b * 8 + g) * 2], rs = norm2[(b * 8 + g) * 2 + 1];
#pragma unroll
      for (int j = 0; j < 8; ++j) {
        int o = O0 + j + 8 * hi;
        float gwv = gw2[o], gbv = gb2[o];
        float mx = -3.0e38f;
#pragma unroll
        for (int t = 0; t < 4; ++t) {
          float h = (acc2[t][j] - mu) * rs * gwv + gbv;
          mx = fmaxf(mx, silu_f(h));
        }
#pragma unroll
        for (int off = 8; off >= 1; off >>= 1) mx = fmaxf(mx, __shfl_xor(mx, off, 32));
        if (l16 == 0) out[(size_t)b * 128 * MCTR + o * MCTR + m] = mx;
      }
    }
  }
}

// ---------------- host launch ----------------
extern "C" void kernel_launch(void* const* d_in, const int* in_sizes, int n_in,
                              void* d_out, int out_size, void* d_ws, size_t ws_size,
                              hipStream_t stream) {
  const float* features = (const float*)d_in[0];
  const float* coords   = (const float*)d_in[1];
  const float* temb     = (const float*)d_in[2];
  const float* w1  = (const float*)d_in[3];
  const float* b1  = (const float*)d_in[4];
  const float* gw1 = (const float*)d_in[5];
  const float* gb1 = (const float*)d_in[6];
  const float* w2  = (const float*)d_in[7];
  const float* b2  = (const float*)d_in[8];
  const float* gw2 = (const float*)d_in[9];
  const float* gb2 = (const float*)d_in[10];

  float* out   = (float*)d_out;                          // (B,128,M)
  float* out_c = out + (size_t)BATCH * 128 * MCTR;       // (B,3,M)
  float* out_t = out_c + (size_t)BATCH * 3 * MCTR;       // (B,64,M)

  char* w = (char*)d_ws;
  int* c_idx = (int*)w;            w += (size_t)BATCH * MCTR * 4;
  int* idx   = (int*)w;            w += (size_t)BATCH * MCTR * KNB * 4;
  _Float16* w1f = (_Float16*)w;    w += 4 * 3 * 32 * 16 * 2;   // 32B-aligned fragment array
  _Float16* w2f = (_Float16*)w;    w += 8 * 2 * 32 * 16 * 2;
  float* ps1   = (float*)w;        w += 64 * MCTR * 4;
  float* pss1  = (float*)w;        w += 64 * MCTR * 4;
  float* ps2   = (float*)w;        w += 64 * MCTR * 4;
  float* pss2  = (float*)w;        w += 64 * MCTR * 4;
  float* norm1 = (float*)w;        w += 64 * 2 * 4;
  float* norm2 = (float*)w;

  prep_kernel<<<24, 256, 0, stream>>>(w1, w2, w1f, w2f);
  fps_kernel<<<BATCH, 1024, 0, stream>>>(coords, c_idx);
  centers_kernel<<<(BATCH * MCTR + 255) / 256, 256, 0, stream>>>(coords, c_idx, out_c);
  ballq_kernel<<<BATCH * MCTR / 8, 256, 0, stream>>>(coords, c_idx, idx);
  temb_kernel<<<BATCH * MCTR, 64, 0, stream>>>(temb, idx, out_t);

  mlp_kernel<1><<<BATCH * MCTR, 128, 0, stream>>>(features, coords, idx, c_idx, w1f, w2f,
      b1, gw1, gb1, b2, gw2, gb2, norm1, norm2, ps1, pss1, ps2, pss2, out);
  finalize_kernel<<<1, 64, 0, stream>>>(ps1, pss1, norm1, 1.f / (8.f * MCTR * KNB));
  mlp_kernel<2><<<BATCH * MCTR, 128, 0, stream>>>(features, coords, idx, c_idx, w1f, w2f,
      b1, gw1, gb1, b2, gw2, gb2, norm1, norm2, ps1, pss1, ps2, pss2, out);
  finalize_kernel<<<1, 64, 0, stream>>>(ps2, pss2, norm2, 1.f / (16.f * MCTR * KNB));
  mlp_kernel<3><<<BATCH * MCTR, 128, 0, stream>>>(features, coords, idx, c_idx, w1f, w2f,
      b1, gw1, gb1, b2, gw2, gb2, norm1, norm2, ps1, pss1, ps2, pss2, out);
}
